// StateTrackerTransformer_3539053051962
// MI455X (gfx1250) — matrix operand order repaired and checked by reading, not compiled, not run on hardware
//
#include <hip/hip_runtime.h>
#include <math.h>
#include <stdint.h>

typedef __attribute__((ext_vector_type(16))) _Float16 v16h;
typedef __attribute__((ext_vector_type(8)))  float    v8f;

#define SEQ    100
#define NPAD   128
#define BATCH  2048
#define DIM    128
#define FREQ   51
#define FPAD   64
#define MPAD   112   // 7*16 tiles for irfft M dimension
#define LAYERS 2
#define LN_EPS 1e-5f

#define USE_ASYNC_LDS 1

// ---------------------------------------------------------------------------
// Async global->LDS 16B copy (GLOBAL_LOAD_ASYNC_TO_LDS_B128, ASYNCcnt).
// LDS byte address = low 32 bits of the generic pointer (flat aperture rule).
// ---------------------------------------------------------------------------
__device__ __forceinline__ void async_copy_b128(const void* gsrc, void* ldst) {
#if USE_ASYNC_LDS
  asm volatile("global_load_async_to_lds_b128 %0, %1, off"
               :: "v"((unsigned int)(uintptr_t)ldst),
                  "v"((unsigned long long)(uintptr_t)gsrc)
               : "memory");
#else
  ((uint4*)ldst)[0] = ((const uint4*)gsrc)[0];
#endif
}

__device__ __forceinline__ void async_copy_wait() {
#if USE_ASYNC_LDS
  asm volatile("s_wait_asynccnt 0x0" ::: "memory");
#endif
}

// ---------------------------------------------------------------------------
// WMMA fragment helpers (wave32; A and B both use contiguous k = 16*(lane/16) + s)
// ---------------------------------------------------------------------------
__device__ __forceinline__ v16h load_a_frag(const _Float16* base, int ld) {
  const int lane = threadIdx.x & 31;
  const _Float16* row = base + (lane & 15) * ld + ((lane >> 4) << 4);
  v16h a;
#pragma unroll
  for (int g = 0; g < 4; ++g) { a[2*g] = row[2*g]; a[2*g+1] = row[2*g+1]; }
#pragma unroll
  for (int g = 0; g < 4; ++g) { a[8+2*g] = row[8+2*g]; a[8+2*g+1] = row[8+2*g+1]; }
  return a;
}

__device__ __forceinline__ v16h load_bT_frag(const _Float16* base, int ld) {
  const int lane = threadIdx.x & 31;
  const _Float16* col = base + (lane & 15) * ld + ((lane >> 4) << 4);
  v16h b;
#pragma unroll
  for (int g = 0; g < 8; ++g) { b[2*g] = col[2*g]; b[2*g+1] = col[2*g+1]; }
  return b;
}

// ---------------------------------------------------------------------------
// Basis matrices: RF/IF (rfft ortho), CR/CI (folded Hermitian irfft ortho).
// ---------------------------------------------------------------------------
__global__ void basis_init_kernel(_Float16* RF, _Float16* IFm,
                                  _Float16* CR, _Float16* CI) {
  const int idx = blockIdx.x * 256 + threadIdx.x;
  const double TWO_PI = 6.283185307179586476925286766559;
  if (idx < FPAD * NPAD) {
    const int f = idx >> 7, n = idx & (NPAD - 1);
    float cr = 0.f, si = 0.f;
    if (f < FREQ && n < SEQ) {
      const double a = TWO_PI * (double)(f * n) / (double)SEQ;
      cr = (float)(cos(a) * 0.1);
      si = (float)(-sin(a) * 0.1);
    }
    RF[idx] = (_Float16)cr;
    IFm[idx] = (_Float16)si;
  }
  if (idx < MPAD * FPAD) {
    const int n = idx >> 6, f = idx & 63;
    float cr = 0.f, ci = 0.f;
    if (n < SEQ && f < FREQ) {
      const double a = TWO_PI * (double)(f * n) / (double)SEQ;
      if (f == 0)              cr = 0.1f;
      else if (f == FREQ - 1)  cr = (n & 1) ? -0.1f : 0.1f;   // Nyquist, (-1)^n
      else { cr = (float)(0.2 * cos(a)); ci = (float)(-0.2 * sin(a)); }
    }
    CR[idx] = (_Float16)cr;
    CI[idx] = (_Float16)ci;
  }
}

// ---------------------------------------------------------------------------
// Fused LayerNorm + forward DFT GEMM per batch b.
//   LN rows of x[:, b, :] -> f16, transposed into LDS sHt[c][n] (n-pad zeroed)
//   Re = RF x H_b, Im = IF x H_b   (WMMA, 64x128 out, K=128)
// ---------------------------------------------------------------------------
__global__ __launch_bounds__(256) void ln_fft_gemm_kernel(
    const float* __restrict__ x, const float* __restrict__ gamma,
    const float* __restrict__ beta, const _Float16* __restrict__ RF,
    const _Float16* __restrict__ IFm, _Float16* __restrict__ ReH,
    _Float16* __restrict__ ImH) {
  __shared__ __align__(16) _Float16 sA[2][FPAD * NPAD];  // RF, IF: 16 KB each
  __shared__ __align__(16) _Float16 sHt[DIM * NPAD];     // H^T [c][n]: 32 KB
  const int b = blockIdx.x, tid = threadIdx.x;

  // async-stage the two DFT matrices (16 KB each, 16 B per lane-issue)
  for (int i = tid; i < FPAD * NPAD / 8; i += 256) {
    async_copy_b128(RF + i * 8, &sA[0][i * 8]);
    async_copy_b128(IFm + i * 8, &sA[1][i * 8]);
  }
  // zero the n in [100,128) pad columns of sHt
  for (int i = tid; i < DIM * (NPAD - SEQ); i += 256) {
    const int c = i / (NPAD - SEQ), n = SEQ + i % (NPAD - SEQ);
    sHt[c * NPAD + n] = (_Float16)0.f;
  }
  // LayerNorm, one wave per sequence row
  const int wave = tid >> 5, lane = tid & 31;
  const float4 g  = ((const float4*)gamma)[lane];
  const float4 be = ((const float4*)beta)[lane];
  for (int n = wave; n < SEQ; n += 8) {
    const float4 v = ((const float4*)(x + ((size_t)n * BATCH + b) * DIM))[lane];
    float s = v.x + v.y + v.z + v.w;
#pragma unroll
    for (int m = 16; m >= 1; m >>= 1) s += __shfl_xor(s, m);
    const float mu = s * (1.f / DIM);
    const float d0 = v.x - mu, d1 = v.y - mu, d2 = v.z - mu, d3 = v.w - mu;
    float ss = d0 * d0 + d1 * d1 + d2 * d2 + d3 * d3;
#pragma unroll
    for (int m = 16; m >= 1; m >>= 1) ss += __shfl_xor(ss, m);
    const float inv = rsqrtf(ss * (1.f / DIM) + LN_EPS);
    const int c0 = lane * 4;
    sHt[(c0 + 0) * NPAD + n] = (_Float16)(d0 * inv * g.x + be.x);
    sHt[(c0 + 1) * NPAD + n] = (_Float16)(d1 * inv * g.y + be.y);
    sHt[(c0 + 2) * NPAD + n] = (_Float16)(d2 * inv * g.z + be.z);
    sHt[(c0 + 3) * NPAD + n] = (_Float16)(d3 * inv * g.w + be.w);
  }
  async_copy_wait();
  __syncthreads();

  for (int j = wave; j < 64; j += 8) {
    const int which = j & 1, tile = j >> 1;
    const int mt = tile >> 3, nt = tile & 7;
    v8f acc = {0.f, 0.f, 0.f, 0.f, 0.f, 0.f, 0.f, 0.f};
#pragma unroll
    for (int kt = 0; kt < 4; ++kt) {
      v16h af = load_a_frag(sA[which] + mt * 16 * NPAD + kt * 32, NPAD);
      v16h bf = load_bT_frag(sHt + nt * 16 * NPAD + kt * 32, NPAD);
      acc = __builtin_amdgcn_wmma_f32_16x16x32_f16(false, af, false, bf,
                                                   (short)0, acc, false, false);
    }
    _Float16* outp = (which ? ImH : ReH) + (size_t)b * FPAD * DIM
                     + mt * 16 * DIM + nt * 16;
    const int n = lane & 15, mo = (lane >> 4) << 3;
#pragma unroll
    for (int r = 0; r < 8; ++r)
      outp[(mo + r) * DIM + n] = (_Float16)acc[r];
  }
}

// ---------------------------------------------------------------------------
// Per-batch spectral energy (from the f16 spectrum, like the reference uses
// one x_fft for both energy and weighting), lower median (rank 25 of 51,
// torch semantics), and norm_e.  Fully deterministic (no atomics).
// ---------------------------------------------------------------------------
__global__ __launch_bounds__(256) void energy_median_kernel(
    const _Float16* __restrict__ ReH, const _Float16* __restrict__ ImH,
    float* __restrict__ norm_e) {
  __shared__ float se[FREQ];
  __shared__ float med;
  const int b = blockIdx.x, tid = threadIdx.x;
  const int wave = tid >> 5, lane = tid & 31;
  const unsigned int* gRe = (const unsigned int*)(ReH + (size_t)b * FPAD * DIM);
  const unsigned int* gIm = (const unsigned int*)(ImH + (size_t)b * FPAD * DIM);
  for (int f = wave; f < FREQ; f += 8) {
    float e = 0.f;
#pragma unroll
    for (int k = 0; k < 2; ++k) {
      union { unsigned int u; _Float16 h[2]; } ur, ui;
      ur.u = gRe[f * 64 + lane * 2 + k];
      ui.u = gIm[f * 64 + lane * 2 + k];
      const float r0 = (float)ur.h[0], r1 = (float)ur.h[1];
      const float i0 = (float)ui.h[0], i1 = (float)ui.h[1];
      e += r0 * r0 + r1 * r1 + i0 * i0 + i1 * i1;
    }
#pragma unroll
    for (int m = 16; m >= 1; m >>= 1) e += __shfl_xor(e, m);
    if (lane == 0) se[f] = e;
  }
  __syncthreads();
  if (tid < FREQ) {
    const float v = se[tid];
    int rank = 0;
    for (int j = 0; j < FREQ; ++j) {
      const float u = se[j];
      rank += (u < v) || (u == v && j < tid);   // stable total order
    }
    if (rank == (FREQ - 1) / 2) med = v;
  }
  __syncthreads();
  if (tid < FREQ) norm_e[b * FPAD + tid] = se[tid] / (med + 1e-6f);
}

// ---------------------------------------------------------------------------
// Global linear-interp quantile (jnp.quantile semantics) over B*51 values.
// Dual 4-pass radix select: k-th and (k+1)-th order statistics share passes.
// ---------------------------------------------------------------------------
__global__ __launch_bounds__(1024) void quantile_kernel(
    const float* __restrict__ norm_e, const float* __restrict__ thr_param,
    int layer, float* __restrict__ thr_out) {
  __shared__ unsigned int hist[2][256];
  __shared__ unsigned int sel_sh[2];
  __shared__ int kk_sh[2];
  const int M = BATCH * FREQ;
  float q = thr_param[layer] * (float)(M - 1);
  if (q < 0.f) q = 0.f;
  if (q > (float)(M - 1)) q = (float)(M - 1);
  int lo = (int)floorf(q);
  if (lo > M - 2) lo = M - 2;
  const float frac = q - (float)lo;

  unsigned int pre0 = 0, pre1 = 0, msk = 0;
  int kk0 = lo, kk1 = lo + 1;
  for (int pass = 0; pass < 4; ++pass) {
    const int shift = 24 - 8 * pass;
    if (threadIdx.x < 512) ((unsigned int*)hist)[threadIdx.x] = 0;
    __syncthreads();
    for (int i = threadIdx.x; i < M; i += 1024) {
      const int bb = i / FREQ, f = i - bb * FREQ;
      unsigned int u = __float_as_uint(norm_e[bb * FPAD + f]);
      u = (u & 0x80000000u) ? ~u : (u | 0x80000000u);
      const unsigned int bin = (u >> shift) & 255u;
      if ((u & msk) == pre0) atomicAdd(&hist[0][bin], 1u);
      if ((u & msk) == pre1) atomicAdd(&hist[1][bin], 1u);
    }
    __syncthreads();
    if (threadIdx.x == 0) {
      for (int s = 0; s < 2; ++s) {
        const int kk = s ? kk1 : kk0;
        const unsigned int pre = s ? pre1 : pre0;
        unsigned int cum = 0; int bin = 0;
        for (; bin < 256; ++bin) {
          const unsigned int h = hist[s][bin];
          if ((unsigned int)kk < cum + h) break;
          cum += h;
        }
        if (bin > 255) bin = 255;
        sel_sh[s] = pre | ((unsigned int)bin << shift);
        kk_sh[s] = kk - (int)cum;
      }
    }
    __syncthreads();
    pre0 = sel_sh[0]; kk0 = kk_sh[0];
    pre1 = sel_sh[1]; kk1 = kk_sh[1];
    msk |= (255u << shift);
    __syncthreads();
  }
  if (threadIdx.x == 0) {
    const float v0 = (pre0 & 0x80000000u) ? __uint_as_float(pre0 & 0x7fffffffu)
                                          : __uint_as_float(~pre0);
    const float v1 = (pre1 & 0x80000000u) ? __uint_as_float(pre1 & 0x7fffffffu)
                                          : __uint_as_float(~pre1);
    thr_out[0] = v0 + frac * (v1 - v0);
  }
}

// ---------------------------------------------------------------------------
// Fused spectral weighting + inverse DFT GEMM per batch b.
//   Staging applies X *= (w + mask*wh) while transposing f16 spectra into LDS.
//   out[n][c] = CR x Re_w + CI x Im_w, residual-added into running state x.
// ---------------------------------------------------------------------------
__global__ __launch_bounds__(256) void wifft_gemm_kernel(
    const _Float16* __restrict__ CR, const _Float16* __restrict__ CI,
    const _Float16* __restrict__ ReH, const _Float16* __restrict__ ImH,
    const float* __restrict__ norm_e, const float* __restrict__ thr_out,
    const float* __restrict__ cw, const float* __restrict__ cwh,
    float* __restrict__ xio) {
  __shared__ __align__(16) _Float16 sCR[MPAD * FPAD];    // 14 KB
  __shared__ __align__(16) _Float16 sCI[MPAD * FPAD];    // 14 KB
  __shared__ __align__(16) _Float16 sReT[DIM * FPAD];    // [c][f]: 16 KB
  __shared__ __align__(16) _Float16 sImT[DIM * FPAD];    // 16 KB
  __shared__ float sMask[FPAD];
  const int b = blockIdx.x, tid = threadIdx.x;

  for (int i = tid; i < MPAD * FPAD / 8; i += 256) {
    async_copy_b128(CR + i * 8, &sCR[i * 8]);
    async_copy_b128(CI + i * 8, &sCI[i * 8]);
  }
  if (tid < FPAD)
    sMask[tid] = (tid < FREQ && norm_e[b * FPAD + tid] < thr_out[0]) ? 1.f : 0.f;
  __syncthreads();   // sMask visible (async CR/CI copies still in flight)

  const unsigned int* gRe = (const unsigned int*)(ReH + (size_t)b * FPAD * DIM);
  const unsigned int* gIm = (const unsigned int*)(ImH + (size_t)b * FPAD * DIM);
  for (int i = tid; i < FPAD * DIM / 2; i += 256) {
    const int f = i >> 6, c2 = (i & 63) << 1;
    union { unsigned int u; _Float16 h[2]; } ur, ui;
    ur.u = gRe[i]; ui.u = gIm[i];
    const float mk = sMask[f];
    const float4 wv = *(const float4*)(cw + c2 * 2);    // w[c2], w[c2+1]
    const float4 wh = *(const float4*)(cwh + c2 * 2);
    const float mr0 = wv.x + mk * wh.x, mi0 = wv.y + mk * wh.y;
    const float mr1 = wv.z + mk * wh.z, mi1 = wv.w + mk * wh.w;
    const float re0 = (float)ur.h[0], im0 = (float)ui.h[0];
    const float re1 = (float)ur.h[1], im1 = (float)ui.h[1];
    sReT[c2 * FPAD + f]       = (_Float16)(re0 * mr0 - im0 * mi0);
    sImT[c2 * FPAD + f]       = (_Float16)(re0 * mi0 + im0 * mr0);
    sReT[(c2 + 1) * FPAD + f] = (_Float16)(re1 * mr1 - im1 * mi1);
    sImT[(c2 + 1) * FPAD + f] = (_Float16)(re1 * mi1 + im1 * mr1);
  }
  async_copy_wait();
  __syncthreads();

  const int wave = tid >> 5, lane = tid & 31;
  for (int j = wave; j < 56; j += 8) {
    const int mt = j >> 3, nt = j & 7;
    v8f acc = {0.f, 0.f, 0.f, 0.f, 0.f, 0.f, 0.f, 0.f};
#pragma unroll
    for (int kt = 0; kt < 2; ++kt) {
      v16h af = load_a_frag(sCR + mt * 16 * FPAD + kt * 32, FPAD);
      v16h bf = load_bT_frag(sReT + nt * 16 * FPAD + kt * 32, FPAD);
      acc = __builtin_amdgcn_wmma_f32_16x16x32_f16(false, af, false, bf,
                                                   (short)0, acc, false, false);
    }
#pragma unroll
    for (int kt = 0; kt < 2; ++kt) {
      v16h af = load_a_frag(sCI + mt * 16 * FPAD + kt * 32, FPAD);
      v16h bf = load_bT_frag(sImT + nt * 16 * FPAD + kt * 32, FPAD);
      acc = __builtin_amdgcn_wmma_f32_16x16x32_f16(false, af, false, bf,
                                                   (short)0, acc, false, false);
    }
    const int n = lane & 15, mo = (lane >> 4) << 3;
    const int c = nt * 16 + n;
#pragma unroll
    for (int r = 0; r < 8; ++r) {
      const int m = mt * 16 + mo + r;
      if (m < SEQ) {
        const size_t o = ((size_t)m * BATCH + b) * DIM + c;
        xio[o] += acc[r];   // residual add into running state
      }
    }
  }
}

// ---------------------------------------------------------------------------
// Host launcher
// ---------------------------------------------------------------------------
extern "C" void kernel_launch(void* const* d_in, const int* in_sizes, int n_in,
                              void* d_out, int out_size, void* d_ws, size_t ws_size,
                              hipStream_t stream) {
  (void)in_sizes; (void)n_in; (void)out_size; (void)ws_size;
  const float* x    = (const float*)d_in[0];
  const float* cw   = (const float*)d_in[1];   // [L, D, 2]
  const float* cwh  = (const float*)d_in[2];   // [L, D, 2]
  const float* thrp = (const float*)d_in[3];   // [L]
  const float* gma  = (const float*)d_in[4];   // [L, D]
  const float* bta  = (const float*)d_in[5];   // [L, D]
  float* xio = (float*)d_out;
  char* ws = (char*)d_ws;

  // Workspace layout (~65 MB total)
  _Float16* RF   = (_Float16*)(ws + 0);                     // 16384 B
  _Float16* IFm  = (_Float16*)(ws + 16384);                 // 16384 B
  _Float16* CR   = (_Float16*)(ws + 32768);                 // 14336 B
  _Float16* CI   = (_Float16*)(ws + 47104);                 // 14336 B
  float*    thrO = (float*)   (ws + 61440);                 // 4 B
  float*    nrme = (float*)   (ws + 65536);                 // 2048*64*4
  _Float16* ReH  = (_Float16*)(ws + 1114112);               // 32 MB
  _Float16* ImH  = (_Float16*)(ws + 1114112 + 33554432);    // 32 MB

  // running state x -> d_out
  hipMemcpyAsync(xio, x, (size_t)SEQ * BATCH * DIM * sizeof(float),
                 hipMemcpyDeviceToDevice, stream);
  basis_init_kernel<<<32, 256, 0, stream>>>(RF, IFm, CR, CI);

  for (int l = 0; l < LAYERS; ++l) {
    ln_fft_gemm_kernel<<<BATCH, 256, 0, stream>>>(
        xio, gma + l * DIM, bta + l * DIM, RF, IFm, ReH, ImH);
    energy_median_kernel<<<BATCH, 256, 0, stream>>>(ReH, ImH, nrme);
    quantile_kernel<<<1, 1024, 0, stream>>>(nrme, thrp, l, thrO);
    wifft_gemm_kernel<<<BATCH, 256, 0, stream>>>(
        CR, CI, ReH, ImH, nrme, thrO, cw + l * DIM * 2, cwh + l * DIM * 2, xio);
  }
}
